// STCrossAttentionBlock_63247688400945
// MI455X (gfx1250) — matrix operand order repaired and checked by reading, not compiled, not run on hardware
//
#include <hip/hip_runtime.h>
#include <hip/hip_bf16.h>
#include <math.h>

#define Bb 8
#define Tt 12
#define Nn 384
#define Dd 128
#define Hh 8
#define HDm 16
#define FFDm 2048
#define KTOP 115
#define M1 (Bb*Tt*Nn)                 /* 36864 */
#define S_ELEMS ((size_t)M1*Dd)       /* 4718592 */

typedef __attribute__((ext_vector_type(16))) __bf16 v16bf;
typedef __attribute__((ext_vector_type(8)))  float  v8f;

union FragBF { v16bf v; unsigned u[8]; };
union AccF   { v8f v; float f[8]; };

__device__ __forceinline__ unsigned short f2bf(float f) {
  unsigned u = __float_as_uint(f);
  u += 0x7FFFu + ((u >> 16) & 1u);     // round-to-nearest-even
  return (unsigned short)(u >> 16);
}
__device__ __forceinline__ unsigned pack2(unsigned short lo, unsigned short hi) {
  return (unsigned)lo | ((unsigned)hi << 16);
}
__device__ __forceinline__ float wredSum(float v){ for(int m=16;m;m>>=1) v += __shfl_xor(v,m); return v; }
__device__ __forceinline__ float wredMax(float v){ for(int m=16;m;m>>=1) v = fmaxf(v,__shfl_xor(v,m)); return v; }
__device__ __forceinline__ int   wredSumI(int v){ for(int m=16;m;m>>=1) v += __shfl_xor(v,m); return v; }

// A-matrix fragment (16xK tile), src row-major [row][k] bf16, p = &src[row*stride + kchunk]
// 16-bit A layout (contiguous): lanes 0-15 hold K=0..15, lanes 16-31 hold K=16..31; koff = (lane>>4)*16
__device__ __forceinline__ void loadFragA(FragBF& f, const unsigned short* p, int lane) {
  int koff = (lane >> 4) * 16;
  #pragma unroll
  for (int v = 0; v < 8; ++v) f.u[v] = *(const unsigned*)(p + koff + 2*v);
}
// B-matrix fragment (Kx16 tile), src stored transposed [col][k] bf16, p = &src[col*stride + kchunk]
// 16-bit B layout (contiguous, same as A): lanes 0-15 hold K=0..15, lanes 16-31 hold K=16..31; koff = (lane>>4)*16
__device__ __forceinline__ void loadFragB(FragBF& f, const unsigned short* p, int lane) {
  int koff = (lane >> 4) * 16;
  #pragma unroll
  for (int v = 0; v < 8; ++v) f.u[v] = *(const unsigned*)(p + koff + 2*v);
}
__device__ __forceinline__ v8f wmma_bf16(const FragBF& a, const FragBF& b, v8f c) {
  return __builtin_amdgcn_wmma_f32_16x16x32_bf16(false, a.v, false, b.v, (short)0, c, false, false);
}

// ---------------------------------------------------------------------------
// Generic 128-wide GEMM: C[M,128] = act(A[M,128] @ W + bias (+resid))
// wtrans=0: W is [128(in),128(out)] ; wtrans=1: W is [128(out),128(in)] (W^T)
// ---------------------------------------------------------------------------
__global__ __launch_bounds__(256) void k_gemm128(
    const float* __restrict__ A, const float* __restrict__ W,
    const float* __restrict__ bias, const float* __restrict__ resid,
    float* __restrict__ C, int wtrans, int relu)
{
  __shared__ unsigned short As[128][34];
  __shared__ unsigned short Ws[128][34];
  const int tid = threadIdx.x, lane = tid & 31, wave = tid >> 5;
  const int mBlock = blockIdx.x * 128;
  const int mRow = wave * 16;
  AccF acc[8];
  #pragma unroll
  for (int j = 0; j < 8; ++j)
    #pragma unroll
    for (int e = 0; e < 8; ++e) acc[j].f[e] = 0.f;

  for (int k0 = 0; k0 < 128; k0 += 32) {
    __syncthreads();
    for (int i = tid; i < 128*32; i += 256) {
      int m = i >> 5, k = i & 31;
      As[m][k] = f2bf(A[(size_t)(mBlock + m) * 128 + k0 + k]);
      Ws[m][k] = f2bf(wtrans ? W[(size_t)m * 128 + (k0 + k)]
                             : W[(size_t)(k0 + k) * 128 + m]);
    }
    __syncthreads();
    FragBF a; loadFragA(a, &As[mRow + (lane & 15)][0], lane);
    #pragma unroll
    for (int j = 0; j < 8; ++j) {
      FragBF b; loadFragB(b, &Ws[j*16 + (lane & 15)][0], lane);
      acc[j].v = wmma_bf16(a, b, acc[j].v);
    }
  }
  #pragma unroll
  for (int j = 0; j < 8; ++j) {
    int col = j * 16 + (lane & 15);
    float bv = bias ? bias[col] : 0.f;
    #pragma unroll
    for (int r = 0; r < 8; ++r) {
      int row = mBlock + mRow + r + ((lane >> 4) * 8);
      float v = acc[j].f[r] + bv;
      if (resid) v += resid[(size_t)row * 128 + col];
      if (relu) v = fmaxf(v, 0.f);
      C[(size_t)row * 128 + col] = v;
    }
  }
}

// ---------------------------------------------------------------------------
// Fused FFN: Out = X + relu(X@W1 + B1) @ W2 + B2    (D=128, F=2048)
// Hidden tensor never hits memory: 64 chunks of 32 hidden units via LDS.
// ---------------------------------------------------------------------------
__global__ __launch_bounds__(256) void k_ffn(
    const float* __restrict__ X, const float* __restrict__ W1, const float* __restrict__ B1,
    const float* __restrict__ W2, const float* __restrict__ B2, float* __restrict__ Out)
{
  __shared__ unsigned short Xs[128][130];   // X block, bf16
  __shared__ unsigned short W1s[32][130];   // W1 chunk transposed [n][k]
  __shared__ unsigned short Hs[128][34];    // hidden chunk [m][k]
  __shared__ unsigned short W2s[128][34];   // W2 chunk transposed [n][k]
  const int tid = threadIdx.x, lane = tid & 31, wave = tid >> 5;
  const int mBlock = blockIdx.x * 128;
  const int mRow = wave * 16;

  for (int i = tid; i < 128*128; i += 256) {
    int m = i >> 7, k = i & 127;
    Xs[m][k] = f2bf(X[(size_t)(mBlock + m) * 128 + k]);
  }
  __syncthreads();
  FragBF xa[4];
  #pragma unroll
  for (int kk = 0; kk < 4; ++kk)
    loadFragA(xa[kk], &Xs[mRow + (lane & 15)][kk*32], lane);

  AccF cacc[8];
  #pragma unroll
  for (int j = 0; j < 8; ++j)
    #pragma unroll
    for (int e = 0; e < 8; ++e) cacc[j].f[e] = 0.f;

  for (int c = 0; c < FFDm/32; ++c) {
    __syncthreads();
    for (int i = tid; i < 32*128; i += 256) {
      int n = i >> 7, k = i & 127;
      W1s[n][k] = f2bf(W1[(size_t)k * FFDm + c*32 + n]);
    }
    for (int i = tid; i < 128*32; i += 256) {
      int n = i >> 5, k = i & 31;
      W2s[n][k] = f2bf(W2[(size_t)(c*32 + k) * 128 + n]);
    }
    __syncthreads();
    // stage 1: H = relu(X@W1 + b1)  (16 rows x 32 cols per wave)
    AccF h[2];
    #pragma unroll
    for (int j = 0; j < 2; ++j)
      #pragma unroll
      for (int e = 0; e < 8; ++e) h[j].f[e] = 0.f;
    #pragma unroll
    for (int kk = 0; kk < 4; ++kk) {
      #pragma unroll
      for (int j = 0; j < 2; ++j) {
        FragBF b; loadFragB(b, &W1s[j*16 + (lane & 15)][kk*32], lane);
        h[j].v = wmma_bf16(xa[kk], b, h[j].v);
      }
    }
    #pragma unroll
    for (int j = 0; j < 2; ++j) {
      int col = j*16 + (lane & 15);
      float bv = B1[c*32 + col];
      #pragma unroll
      for (int r = 0; r < 8; ++r) {
        int row = mRow + r + ((lane >> 4) * 8);
        Hs[row][col] = f2bf(fmaxf(h[j].f[r] + bv, 0.f));
      }
    }
    __syncthreads();
    // stage 2: C += H @ W2
    FragBF ha; loadFragA(ha, &Hs[mRow + (lane & 15)][0], lane);
    #pragma unroll
    for (int j = 0; j < 8; ++j) {
      FragBF b; loadFragB(b, &W2s[j*16 + (lane & 15)][0], lane);
      cacc[j].v = wmma_bf16(ha, b, cacc[j].v);
    }
  }
  #pragma unroll
  for (int j = 0; j < 8; ++j) {
    int col = j * 16 + (lane & 15);
    float bv = B2[col];
    #pragma unroll
    for (int r = 0; r < 8; ++r) {
      int row = mBlock + mRow + r + ((lane >> 4) * 8);
      Out[(size_t)row * 128 + col] = cacc[j].f[r] + bv + X[(size_t)row * 128 + col];
    }
  }
}

// ---------------------------------------------------------------------------
// Spatial attention with adp mask + softmax + exact top-K(115) + P@Vt apply.
// Grid: (N/16, B*T, H). 16 query rows per workgroup. Dynamic LDS.
// ---------------------------------------------------------------------------
__global__ __launch_bounds__(256) void k_spatial(
    const float* __restrict__ sq, const float* __restrict__ sk,
    const float* __restrict__ tv, const float* __restrict__ adp,
    float* __restrict__ outp)
{
  extern __shared__ char smem[];
  unsigned short* Kb = (unsigned short*)smem;              // [384][16] bf16
  unsigned short* Vb = (unsigned short*)(smem + 12288);    // [384][16] bf16
  float*          Sc = (float*)(smem + 24576);             // [16][400] f32
  unsigned short* Pb = (unsigned short*)(smem + 50176);    // [16][392] bf16
  float*          Pt = (float*)(smem + 62720);             // [8][256] partials

  const int tid = threadIdx.x, lane = tid & 31, wave = tid >> 5;
  const int r0 = blockIdx.x * 16;
  const int bt = blockIdx.y;
  const int b = bt / Tt, t = bt % Tt;
  const int h = blockIdx.z;

  for (int i = tid; i < Nn*HDm; i += 256) {
    int m = i >> 4, hd = i & 15;
    Kb[i] = f2bf(sk[((size_t)bt * Nn + m) * Dd + h*HDm + hd]);
    Vb[i] = f2bf(tv[((size_t)(b*Nn + m) * Tt + t) * Dd + h*HDm + hd]);   // t_v_s
  }
  __syncthreads();

  // Q fragment (K-dim 16 zero-padded to 32)
  FragBF qa;
  {
    int m = lane & 15, hiLane = lane >> 4;
    const float* qp = sq + ((size_t)bt * Nn + r0 + m) * Dd + h*HDm;
    #pragma unroll
    for (int v = 0; v < 8; ++v)
      qa.u[v] = hiLane ? 0u : pack2(f2bf(qp[2*v]), f2bf(qp[2*v + 1]));
  }
  // scores: 24 column tiles, 3 per wave
  for (int jj = 0; jj < 3; ++jj) {
    int cb = (wave*3 + jj) * 16;
    FragBF kf;
    {
      const unsigned short* kp = Kb + (size_t)(cb + (lane & 15)) * HDm;
      int hiLane = lane >> 4;                // lanes>=16 hold K=16..31 -> zero pad
      #pragma unroll
      for (int v = 0; v < 8; ++v) {
        unsigned u = *(const unsigned*)(kp + 2*v);
        kf.u[v] = hiLane ? 0u : u;
      }
    }
    AccF a;
    #pragma unroll
    for (int e = 0; e < 8; ++e) a.f[e] = 0.f;
    a.v = wmma_bf16(qa, kf, a.v);
    #pragma unroll
    for (int r = 0; r < 8; ++r) {
      int row = r + ((lane >> 4) * 8);
      int col = cb + (lane & 15);
      Sc[row*400 + col] = a.f[r] * 0.25f * adp[(size_t)(r0 + row) * Nn + col];
    }
  }
  __syncthreads();

  // softmax + exact top-K per row (2 rows per wave, 12 cols per lane)
  for (int rr = 0; rr < 2; ++rr) {
    int row = wave*2 + rr;
    float pv[12]; unsigned pbits[12];
    float mx = -1e30f;
    #pragma unroll
    for (int i = 0; i < 12; ++i) { pv[i] = Sc[row*400 + lane + 32*i]; mx = fmaxf(mx, pv[i]); }
    mx = wredMax(mx);
    float s = 0.f;
    #pragma unroll
    for (int i = 0; i < 12; ++i) { pv[i] = __expf(pv[i] - mx); s += pv[i]; }
    s = wredSum(s);
    float inv = 1.f / s;
    #pragma unroll
    for (int i = 0; i < 12; ++i) { pv[i] *= inv; pbits[i] = __float_as_uint(pv[i]); }
    // binary search on float bits for the K-th largest value (probs >= 0)
    unsigned th = 0u;
    for (int bit = 30; bit >= 0; --bit) {
      unsigned cand = th | (1u << bit);
      int cnt = 0;
      #pragma unroll
      for (int i = 0; i < 12; ++i) cnt += (pbits[i] >= cand) ? 1 : 0;
      cnt = wredSumI(cnt);
      if (cnt >= KTOP) th = cand;
    }
    #pragma unroll
    for (int i = 0; i < 12; ++i)
      Pb[row*392 + lane + 32*i] = f2bf(pbits[i] >= th ? pv[i] : 0.f);
  }
  __syncthreads();

  // apply: out(16x16) = P(16x384) @ V(384x16); 12 K-steps split over 8 waves
  AccF o;
  #pragma unroll
  for (int e = 0; e < 8; ++e) o.f[e] = 0.f;
  int nsteps = (wave < 4) ? 2 : 1;
  for (int ss = 0; ss < nsteps; ++ss) {
    int kb0 = (wave + ss*8) * 32;
    FragBF pa; loadFragA(pa, Pb + (size_t)(lane & 15)*392 + kb0, lane);
    FragBF vf;
    {
      int n = lane & 15, koff = (lane >> 4) * 16;
      #pragma unroll
      for (int v = 0; v < 8; ++v) {
        int k = kb0 + koff + 2*v;
        vf.u[v] = pack2(Vb[(size_t)k*HDm + n], Vb[(size_t)(k+1)*HDm + n]);
      }
    }
    o.v = wmma_bf16(pa, vf, o.v);
  }
  {
    int colL = lane & 15;
    #pragma unroll
    for (int r = 0; r < 8; ++r) {
      int row = r + ((lane >> 4) * 8);
      Pt[wave*256 + row*16 + colL] = o.f[r];
    }
  }
  __syncthreads();
  {
    int row = tid >> 4, col = tid & 15;
    float sum = 0.f;
    #pragma unroll
    for (int w = 0; w < 8; ++w) sum += Pt[w*256 + row*16 + col];
    outp[((size_t)bt * Nn + r0 + row) * Dd + h*HDm + col] = sum;
  }
}

// ---------------------------------------------------------------------------
// Temporal attention (12x12, negligible FLOPs): wave per (h,b,n).
// tq/tk in [B,N,T,D]; sv in [B,T,N,D]; output in [B,T,N,D].
// ---------------------------------------------------------------------------
__global__ __launch_bounds__(256) void k_temporal(
    const float* __restrict__ tq, const float* __restrict__ tk,
    const float* __restrict__ sv, float* __restrict__ outp)
{
  const int lane = threadIdx.x & 31, h = threadIdx.x >> 5;
  const int bn = blockIdx.x;
  const int b = bn / Nn, n = bn % Nn;
  const size_t baseT = ((size_t)bn * Tt) * Dd + h*HDm;
  float kreg[16];
  {
    int srow = (lane < Tt) ? lane : 0;
    #pragma unroll
    for (int d = 0; d < 16; ++d) kreg[d] = tk[baseT + (size_t)srow*Dd + d];
  }
  float svv[12];
  {
    int d = (lane < 16) ? lane : 0;
    #pragma unroll
    for (int s = 0; s < Tt; ++s)
      svv[s] = sv[((size_t)(b*Tt + s)*Nn + n)*Dd + h*HDm + d];
  }
  for (int t = 0; t < Tt; ++t) {
    float sc = 0.f;
    #pragma unroll
    for (int d = 0; d < 16; ++d) sc += tq[baseT + (size_t)t*Dd + d] * kreg[d];
    sc *= 0.25f;
    float v = (lane < Tt) ? sc : -1e30f;
    float mx = wredMax(v);
    float e = (lane < Tt) ? __expf(v - mx) : 0.f;
    float ssum = wredSum(e);
    float p = e / ssum;
    float o = 0.f;
    #pragma unroll
    for (int s = 0; s < Tt; ++s) o += __shfl(p, s) * svv[s];
    if (lane < 16)
      outp[((size_t)(b*Tt + t)*Nn + n)*Dd + h*HDm + lane] = o;
  }
}

// ---------------------------------------------------------------------------
// LayerNorm over last dim (128): one wave per row.
// ---------------------------------------------------------------------------
__global__ __launch_bounds__(256) void k_ln(
    const float* __restrict__ X, const float* __restrict__ g,
    const float* __restrict__ be, float* __restrict__ Out)
{
  const int lane = threadIdx.x & 31, wave = threadIdx.x >> 5;
  const size_t row = (size_t)blockIdx.x * 8 + wave;
  const float* xp = X + row * Dd + lane*4;
  float v0 = xp[0], v1 = xp[1], v2 = xp[2], v3 = xp[3];
  float m = wredSum(v0+v1+v2+v3) * (1.f/128.f);
  float d0=v0-m, d1=v1-m, d2=v2-m, d3=v3-m;
  float var = wredSum(d0*d0+d1*d1+d2*d2+d3*d3) * (1.f/128.f);
  float rs = rsqrtf(var + 1e-5f);
  const float* gp = g + lane*4; const float* bp = be + lane*4;
  float* op = Out + row * Dd + lane*4;
  op[0]=d0*rs*gp[0]+bp[0]; op[1]=d1*rs*gp[1]+bp[1];
  op[2]=d2*rs*gp[2]+bp[2]; op[3]=d3*rs*gp[3]+bp[3];
}

// temporal moving average (kernel 5, replicate pad), x[B,T,N,D] -> tin[B,N,T,D]
__global__ void k_tin(const float* __restrict__ x, float* __restrict__ tin)
{
  size_t idx = (size_t)blockIdx.x * 256 + threadIdx.x;
  int d = (int)(idx & 127);
  size_t r = idx >> 7;
  int t = (int)(r % Tt);
  size_t r2 = r / Tt;
  int n = (int)(r2 % Nn);
  int b = (int)(r2 / Nn);
  float s = 0.f;
  #pragma unroll
  for (int dt = -2; dt <= 2; ++dt) {
    int tc = t + dt; tc = tc < 0 ? 0 : (tc > Tt-1 ? Tt-1 : tc);
    s += x[((size_t)(b*Tt + tc)*Nn + n)*Dd + d];
  }
  tin[idx] = s * 0.2f;
}

// gated fusion: out = sigmoid(g1+g2)*sA + (1-sigmoid)*tA
__global__ void k_gate(const float* __restrict__ g1, const float* __restrict__ g2,
                       const float* __restrict__ sA, const float* __restrict__ tA,
                       float* __restrict__ out)
{
  size_t i = (size_t)blockIdx.x * 256 + threadIdx.x;
  float z = 1.f / (1.f + __expf(-(g1[i] + g2[i])));
  out[i] = z * sA[i] + (1.f - z) * tA[i];
}

// ---------------------------------------------------------------------------
extern "C" void kernel_launch(void* const* d_in, const int* in_sizes, int n_in,
                              void* d_out, int out_size, void* d_ws, size_t ws_size,
                              hipStream_t stream)
{
  (void)in_sizes; (void)n_in; (void)out_size; (void)ws_size;
  const float* x   = (const float*)d_in[0];
  const float* adp = (const float*)d_in[1];
  const float* sc_w = (const float*)d_in[2];  const float* sc_b = (const float*)d_in[3];
  const float* sq_w = (const float*)d_in[4];  const float* sq_b = (const float*)d_in[5];
  const float* sk_w = (const float*)d_in[6];  const float* sk_b = (const float*)d_in[7];
  const float* sv_w = (const float*)d_in[8];  const float* sv_b = (const float*)d_in[9];
  const float* tq_w = (const float*)d_in[10]; const float* tq_b = (const float*)d_in[11];
  const float* tk_w = (const float*)d_in[12]; const float* tk_b = (const float*)d_in[13];
  const float* tv_w = (const float*)d_in[14]; const float* tv_b = (const float*)d_in[15];
  const float* sp_w = (const float*)d_in[16]; const float* sp_b = (const float*)d_in[17];
  const float* tp_w = (const float*)d_in[18]; const float* tp_b = (const float*)d_in[19];
  const float* fs_w = (const float*)d_in[20]; const float* fs_b = (const float*)d_in[21];
  const float* ft_w = (const float*)d_in[22]; const float* ft_b = (const float*)d_in[23];
  const float* sff1_w = (const float*)d_in[24]; const float* sff1_b = (const float*)d_in[25];
  const float* sff2_w = (const float*)d_in[26]; const float* sff2_b = (const float*)d_in[27];
  const float* tff1_w = (const float*)d_in[28]; const float* tff1_b = (const float*)d_in[29];
  const float* tff2_w = (const float*)d_in[30]; const float* tff2_b = (const float*)d_in[31];
  const float* ff1_w  = (const float*)d_in[32]; const float* ff1_b  = (const float*)d_in[33];
  const float* ff2_w  = (const float*)d_in[34]; const float* ff2_b  = (const float*)d_in[35];
  const float* sln1_g = (const float*)d_in[36]; const float* sln1_b = (const float*)d_in[37];
  const float* tln1_g = (const float*)d_in[38]; const float* tln1_b = (const float*)d_in[39];
  const float* sln2_g = (const float*)d_in[40]; const float* sln2_b = (const float*)d_in[41];
  const float* tln2_g = (const float*)d_in[42]; const float* tln2_b = (const float*)d_in[43];
  const float* ln_g   = (const float*)d_in[44]; const float* ln_b   = (const float*)d_in[45];
  const float* fc_w0  = (const float*)d_in[46];
  const float* fc_w1  = (const float*)d_in[47];
  const float* fc_b   = (const float*)d_in[48];

  float* ws = (float*)d_ws;
  float* p0 = ws + 0*S_ELEMS;   // s_in / s_ln1
  float* p1 = ws + 1*S_ELEMS;   // t_in / t_ln1
  float* p2 = ws + 2*S_ELEMS;   // sq / scratch
  float* p3 = ws + 3*S_ELEMS;   // sk / scratch
  float* p4 = ws + 4*S_ELEMS;   // sv / s_out2
  float* p5 = ws + 5*S_ELEMS;   // tq / t_out2
  float* p6 = ws + 6*S_ELEMS;   // tk / fs / fused
  float* p7 = ws + 7*S_ELEMS;   // tv / ft / scratch
  float* p8 = ws + 8*S_ELEMS;   // s_att
  float* p9 = ws + 9*S_ELEMS;   // t_att
  float* out = (float*)d_out;

  dim3 blk(256);
  const int gE = (int)(S_ELEMS / 256);  // 18432
  const int gG = M1 / 128;              // 288
  const int gL = M1 / 8;                // 4608
  const unsigned SPAT_SMEM = 70912u;

  // temporal moving-average input (to [B,N,T,D]) and spatial 1x1 conv
  k_tin<<<gE, blk, 0, stream>>>(x, p1);
  k_gemm128<<<gG, blk, 0, stream>>>(x,  sc_w, sc_b, nullptr, p0, 0, 0);

  // QKV projections
  k_gemm128<<<gG, blk, 0, stream>>>(p0, sq_w, sq_b, nullptr, p2, 0, 0);
  k_gemm128<<<gG, blk, 0, stream>>>(p0, sk_w, sk_b, nullptr, p3, 0, 0);
  k_gemm128<<<gG, blk, 0, stream>>>(p0, sv_w, sv_b, nullptr, p4, 0, 0);
  k_gemm128<<<gG, blk, 0, stream>>>(p1, tq_w, tq_b, nullptr, p5, 0, 0);
  k_gemm128<<<gG, blk, 0, stream>>>(p1, tk_w, tk_b, nullptr, p6, 0, 0);
  k_gemm128<<<gG, blk, 0, stream>>>(p1, tv_w, tv_b, nullptr, p7, 0, 0);

  // cross attentions
  k_spatial<<<dim3(Nn/16, Bb*Tt, Hh), blk, SPAT_SMEM, stream>>>(p2, p3, p7, adp, p8);
  k_temporal<<<Bb*Nn, blk, 0, stream>>>(p5, p6, p4, p9);

  // output projections + residual + LN
  k_gemm128<<<gG, blk, 0, stream>>>(p8, sp_w, sp_b, x, p2, 0, 0);
  k_ln<<<gL, blk, 0, stream>>>(p2, sln1_g, sln1_b, p0);
  k_gemm128<<<gG, blk, 0, stream>>>(p9, tp_w, tp_b, x, p3, 0, 0);
  k_ln<<<gL, blk, 0, stream>>>(p3, tln1_g, tln1_b, p1);

  // branch FFNs (fused, residual inside) + LN
  k_ffn<<<gG, blk, 0, stream>>>(p0, sff1_w, sff1_b, sff2_w, sff2_b, p2);
  k_ln<<<gL, blk, 0, stream>>>(p2, sln2_g, sln2_b, p4);   // s_out2
  k_ffn<<<gG, blk, 0, stream>>>(p1, tff1_w, tff1_b, tff2_w, tff2_b, p3);
  k_ln<<<gL, blk, 0, stream>>>(p3, tln2_g, tln2_b, p5);   // t_out2

  // gated fusion:  z = sigmoid(fs@fc_w0^T + ft@fc_w1^T + fc_b)
  k_gemm128<<<gG, blk, 0, stream>>>(p4, fs_w, fs_b, nullptr, p6, 0, 0);
  k_gemm128<<<gG, blk, 0, stream>>>(p5, ft_w, ft_b, nullptr, p7, 0, 0);
  k_gemm128<<<gG, blk, 0, stream>>>(p6, fc_w0, fc_b, nullptr, p2, 1, 0);   // W^T
  k_gemm128<<<gG, blk, 0, stream>>>(p7, fc_w1, nullptr, nullptr, p3, 1, 0);
  k_gate<<<gE, blk, 0, stream>>>(p2, p3, p4, p5, p6);

  // final FFN + residual + LN -> output
  k_ffn<<<gG, blk, 0, stream>>>(p6, ff1_w, ff1_b, ff2_w, ff2_b, p7);
  k_ln<<<gL, blk, 0, stream>>>(p7, ln_g, ln_b, out);
}
